// GaussianNewLifter_34394098107047
// MI455X (gfx1250) — matrix operand order repaired and checked by reading, not compiled, not run on hardware
//
#include <hip/hip_runtime.h>
#include <hip/hip_bf16.h>

typedef _Float16 half16 __attribute__((ext_vector_type(16)));
typedef float    float8 __attribute__((ext_vector_type(8)));

union AFrag { half16 h; unsigned u[8]; };
union CFrag { float8 f; float x[8]; };

#define N_ANCHOR        409600
#define ANCH_PER_BLOCK  128
#define THREADS         256

__device__ __forceinline__ float sigmoid_clip(float v) {
    float s = 1.0f / (1.0f + __expf(-v));
    return fminf(fmaxf(s, 0.01f), 0.99f);
}
__device__ __forceinline__ float inv_sigmoid(float x) {
    x = fminf(fmaxf(x, 0.01f), 0.99f);
    return __logf(x / (1.0f - x));
}
__device__ __forceinline__ unsigned pk2(float a, float b) {
    union { _Float16 h[2]; unsigned u; } t;
    t.h[0] = (_Float16)a; t.h[1] = (_Float16)b; return t.u;
}

__global__ __launch_bounds__(THREADS)
void GaussianNewLifter_kernel(
    const float* __restrict__ anchor,
    const float* __restrict__ depthnet,
    const float* __restrict__ depthgt,
    const float* __restrict__ w2c,
    const float* __restrict__ cam_k,
    const float* __restrict__ vox_origin,
    const float* __restrict__ scene_size,
    const float* __restrict__ cvr,
    const float* __restrict__ W1, const float* __restrict__ b1,
    const float* __restrict__ W2, const float* __restrict__ b2,
    const float* __restrict__ W3, const float* __restrict__ b3,
    const float* __restrict__ Wi, const float* __restrict__ bi,
    const int*   __restrict__ flag1,
    const int*   __restrict__ flag2,
    float* __restrict__ outA,   // N*24
    float* __restrict__ outI)   // N*256
{
    // ---- LDS: fragment-ready f16 weights + staged activations ----
    __shared__ unsigned sW2[16 * 256];   // 16 KB : (kb0..1, nb0..7)
    __shared__ unsigned sW3[64 * 256];   // 64 KB : (kb0..3, nb0..15)
    __shared__ unsigned sWi[16 * 256];   // 16 KB : (nb0..15), K padded 24->32
    __shared__ _Float16 sH1[128 * 64];   // 16 KB
    __shared__ _Float16 sH2[128 * 128];  // 32 KB
    __shared__ _Float16 sAO[128 * 32];   //  8 KB (24 cols + 8 zero pad)
    __shared__ float    sDF[128 * 2];    //  1 KB

    const int tid = threadIdx.x;

    // ================= Phase 0: pre-pack weight B-fragments =================
    // B fragment layout (32x16 f16): lane l -> col n = l&15, h = l>>4;
    // VGPR v packs K = (v>>2)*16 + h*8 + 2(v&3) (lo 16b) and that K+1 (hi 16b).
    for (int i = tid; i < 16 * 256; i += THREADS) {            // W2 (64x128)
        int f = i >> 8, r = i & 255;
        int lane = r >> 3, v = r & 7;
        int kb = f >> 3, nb = f & 7;
        int n = lane & 15;
        int k = kb * 32 + ((v >> 2) << 4) + ((lane >> 4) << 3) + 2 * (v & 3);
        int col = nb * 16 + n;
        sW2[i] = pk2(W2[k * 128 + col], W2[(k + 1) * 128 + col]);
    }
    for (int i = tid; i < 64 * 256; i += THREADS) {            // W3 (128x256)
        int f = i >> 8, r = i & 255;
        int lane = r >> 3, v = r & 7;
        int kb = f >> 4, nb = f & 15;
        int n = lane & 15;
        int k = kb * 32 + ((v >> 2) << 4) + ((lane >> 4) << 3) + 2 * (v & 3);
        int col = nb * 16 + n;
        sW3[i] = pk2(W3[k * 256 + col], W3[(k + 1) * 256 + col]);
    }
    for (int i = tid; i < 16 * 256; i += THREADS) {            // Wi (24x256, pad)
        int f = i >> 8, r = i & 255;
        int lane = r >> 3, v = r & 7;
        int nb = f;
        int n = lane & 15;
        int k = ((v >> 2) << 4) + ((lane >> 4) << 3) + 2 * (v & 3);
        int col = nb * 16 + n;
        float a0 = (k     < 24) ? Wi[k * 256 + col]       : 0.0f;
        float a1 = (k + 1 < 24) ? Wi[(k + 1) * 256 + col] : 0.0f;
        sWi[i] = pk2(a0, a1);
    }

    // camera quaternion (uniform across threads)
    float m00 = w2c[0], m01 = w2c[1], m02 = w2c[2];
    float m10 = w2c[4], m11 = w2c[5], m12 = w2c[6];
    float m20 = w2c[8], m21 = w2c[9], m22 = w2c[10];
    float tr = m00 + m11 + m22;
    float qw, qx, qy, qz;
    if (tr > 0.f) {
        float s = sqrtf(fmaxf(tr + 1.f, 1e-8f)) * 2.f;
        qw = 0.25f * s; qx = (m21 - m12) / s; qy = (m02 - m20) / s; qz = (m10 - m01) / s;
    } else if (m00 > m11 && m00 > m22) {
        float s = sqrtf(fmaxf(1.f + m00 - m11 - m22, 1e-8f)) * 2.f;
        qw = (m21 - m12) / s; qx = 0.25f * s; qy = (m01 + m10) / s; qz = (m02 + m20) / s;
    } else if (m11 > m22) {
        float s = sqrtf(fmaxf(1.f + m11 - m00 - m22, 1e-8f)) * 2.f;
        qw = (m02 - m20) / s; qx = (m01 + m10) / s; qy = 0.25f * s; qz = (m12 + m21) / s;
    } else {
        float s = sqrtf(fmaxf(1.f + m22 - m00 - m11, 1e-8f)) * 2.f;
        qw = (m10 - m01) / s; qx = (m02 + m20) / s; qy = (m12 + m21) / s; qz = 0.25f * s;
    }
    float qn = fmaxf(sqrtf(qw * qw + qx * qx + qy * qy + qz * qz), 1e-8f);
    qw /= qn; qx /= qn; qy /= qn; qz /= qn;

    // ================= Phase 1: per-anchor geometry =================
    if (tid < ANCH_PER_BLOCK) {
        int gi = blockIdx.x * ANCH_PER_BLOCK + tid;
        const float* ar = anchor + (size_t)gi * 24;
        float sx = sigmoid_clip(ar[0]);
        float sy = sigmoid_clip(ar[1]);
        float sz = sigmoid_clip(ar[2]);
        float xw = sx * scene_size[0] + vox_origin[0];
        float yw = sy * scene_size[1] + vox_origin[1];
        float zw = sz * scene_size[2] + vox_origin[2];
        float c0 = w2c[0] * xw + w2c[1] * yw + w2c[2]  * zw + w2c[3];
        float c1 = w2c[4] * xw + w2c[5] * yw + w2c[6]  * zw + w2c[7];
        float c2 = w2c[8] * xw + w2c[9] * yw + w2c[10] * zw + w2c[11];
        float fx = cam_k[0], fy = cam_k[4], cxp = cam_k[2], cyp = cam_k[5];
        int px = (int)fminf(fmaxf(fx * c0 / c2 + cxp, 0.f), 639.f);
        int py = (int)fminf(fmaxf(fy * c1 / c2 + cyp, 0.f), 479.f);
        const float* zm = (flag1[0] && flag2[0]) ? depthnet : depthgt;
        float depth = zm[py * 640 + px];
        sDF[tid * 2 + 0] = depth;
        sDF[tid * 2 + 1] = c2;

        float ao[24];
        float c3v[3] = { c0, c1, c2 };
        #pragma unroll
        for (int d = 0; d < 3; ++d) {
            float lo = cvr[d], hi = cvr[3 + d];
            float pc = fminf(fmaxf(c3v[d], lo), hi);
            pc = (pc - lo) / (hi - lo);
            pc = fminf(fmaxf(pc, 0.001f), 0.999f);
            ao[d] = inv_sigmoid(pc);
        }
        ao[3] = ar[3]; ao[4] = ar[4]; ao[5] = ar[5];
        float w2_ = ar[6], x2_ = ar[7], y2_ = ar[8], z2_ = ar[9];
        float ow = qw * w2_ - qx * x2_ - qy * y2_ - qz * z2_;
        float ox = qw * x2_ + qx * w2_ + qy * z2_ - qz * y2_;
        float oy = qw * y2_ - qx * z2_ + qy * w2_ + qz * x2_;
        float oz = qw * z2_ + qx * y2_ - qy * x2_ + qz * w2_;
        float on = fmaxf(sqrtf(ow * ow + ox * ox + oy * oy + oz * oz), 1e-8f);
        ao[6] = ow / on; ao[7] = ox / on; ao[8] = oy / on; ao[9] = oz / on;
        #pragma unroll
        for (int c = 10; c < 24; ++c) ao[c] = ar[c];

        float* oA = outA + (size_t)gi * 24;
        #pragma unroll
        for (int c = 0; c < 24; ++c) oA[c] = ao[c];
        _Float16* aoh = sAO + tid * 32;
        #pragma unroll
        for (int c = 0; c < 24; ++c) aoh[c] = (_Float16)ao[c];
        #pragma unroll
        for (int c = 24; c < 32; ++c) aoh[c] = (_Float16)0.f;
    }
    __syncthreads();

    // ================= Phase 2: tiny layer 2->64 (scalar) =================
    const int wave  = tid >> 5;
    const int lane  = tid & 31;
    const int mloc  = lane & 15;           // row within wave's 16-anchor tile
    const int khalf = lane >> 4;
    {
        int aLoc = wave * 16 + mloc;
        int j0 = khalf * 32;
        float d0 = sDF[aLoc * 2 + 0], d1 = sDF[aLoc * 2 + 1];
        for (int j = j0; j < j0 + 32; ++j) {
            float h = fmaxf(d0 * W1[j] + d1 * W1[64 + j] + b1[j], 0.f);
            sH1[aLoc * 64 + j] = (_Float16)h;
        }
    }
    __syncthreads();

    // A-fragment loader: 16-bit A 16x32 layout. Lane l: row = l&15, khalf = l>>4;
    // v<4 -> K = khalf*8+2v ; v>=4 -> K = 16+khalf*8+2(v-4). Pairs are K-contiguous.
    // (dword index = K/2)

    // ================= Phase 3: 64 -> 128 WMMA, relu =================
    {
        const unsigned* h1row = (const unsigned*)(sH1 + (wave * 16 + mloc) * 64);
        AFrag a1[2];
        #pragma unroll
        for (int kb = 0; kb < 2; ++kb) {
            const unsigned* p = h1row + kb * 16;
            int o = khalf * 4;
            #pragma unroll
            for (int j = 0; j < 4; ++j) { a1[kb].u[j] = p[o + j]; a1[kb].u[4 + j] = p[8 + o + j]; }
        }
        #pragma unroll
        for (int nb = 0; nb < 8; ++nb) {
            CFrag c;
            #pragma unroll
            for (int j = 0; j < 8; ++j) c.x[j] = 0.f;
            #pragma unroll
            for (int kb = 0; kb < 2; ++kb) {
                AFrag b;
                const unsigned* q = sW2 + (kb * 8 + nb) * 256 + lane * 8;
                #pragma unroll
                for (int j = 0; j < 8; ++j) b.u[j] = q[j];
                c.f = __builtin_amdgcn_wmma_f32_16x16x32_f16(
                        false, a1[kb].h, false, b.h, (short)0, c.f, false, false);
            }
            int n = nb * 16 + mloc;
            float bias = b2[n];
            int mrow0 = wave * 16 + 8 * khalf;
            #pragma unroll
            for (int j = 0; j < 8; ++j) {
                float v = fmaxf(c.x[j] + bias, 0.f);
                sH2[(mrow0 + j) * 128 + n] = (_Float16)v;
            }
        }
    }
    __syncthreads();

    // ===== Phase 4: fused instance = h2@W3 + anchor_out@Wi + (b3+bi) =====
    {
        const unsigned* h2row = (const unsigned*)(sH2 + (wave * 16 + mloc) * 128);
        AFrag a2[4];
        #pragma unroll
        for (int kb = 0; kb < 4; ++kb) {
            const unsigned* p = h2row + kb * 16;
            int o = khalf * 4;
            #pragma unroll
            for (int j = 0; j < 4; ++j) { a2[kb].u[j] = p[o + j]; a2[kb].u[4 + j] = p[8 + o + j]; }
        }
        AFrag aao;
        {
            const unsigned* p = (const unsigned*)(sAO + (wave * 16 + mloc) * 32);
            int o = khalf * 4;
            #pragma unroll
            for (int j = 0; j < 4; ++j) { aao.u[j] = p[o + j]; aao.u[4 + j] = p[8 + o + j]; }
        }
        int gBase = blockIdx.x * ANCH_PER_BLOCK + wave * 16;
        #pragma unroll
        for (int nb = 0; nb < 16; ++nb) {
            CFrag c;
            #pragma unroll
            for (int j = 0; j < 8; ++j) c.x[j] = 0.f;
            #pragma unroll
            for (int kb = 0; kb < 4; ++kb) {
                AFrag b;
                const unsigned* q = sW3 + (kb * 16 + nb) * 256 + lane * 8;
                #pragma unroll
                for (int j = 0; j < 8; ++j) b.u[j] = q[j];
                c.f = __builtin_amdgcn_wmma_f32_16x16x32_f16(
                        false, a2[kb].h, false, b.h, (short)0, c.f, false, false);
            }
            {
                AFrag b;
                const unsigned* q = sWi + nb * 256 + lane * 8;
                #pragma unroll
                for (int j = 0; j < 8; ++j) b.u[j] = q[j];
                c.f = __builtin_amdgcn_wmma_f32_16x16x32_f16(
                        false, aao.h, false, b.h, (short)0, c.f, false, false);
            }
            int n = nb * 16 + mloc;
            float bias = b3[n] + bi[n];
            #pragma unroll
            for (int j = 0; j < 8; ++j) {
                int row = gBase + 8 * khalf + j;
                outI[(size_t)row * 256 + n] = c.x[j] + bias;
            }
        }
    }
}

extern "C" void kernel_launch(void* const* d_in, const int* in_sizes, int n_in,
                              void* d_out, int out_size, void* d_ws, size_t ws_size,
                              hipStream_t stream) {
    const float* anchor     = (const float*)d_in[0];
    const float* depthnet   = (const float*)d_in[1];
    const float* depthgt    = (const float*)d_in[2];
    // d_in[3] = img_feats (only batch size used by reference; not needed)
    const float* w2c        = (const float*)d_in[4];
    const float* cam_k      = (const float*)d_in[5];
    const float* vox_origin = (const float*)d_in[6];
    const float* scene_size = (const float*)d_in[7];
    const float* cvr        = (const float*)d_in[8];
    const float* W1 = (const float*)d_in[9];
    const float* b1 = (const float*)d_in[10];
    const float* W2 = (const float*)d_in[11];
    const float* b2 = (const float*)d_in[12];
    const float* W3 = (const float*)d_in[13];
    const float* b3 = (const float*)d_in[14];
    const float* Wi = (const float*)d_in[15];
    const float* bi = (const float*)d_in[16];
    const int* f1 = (const int*)d_in[17];
    const int* f2 = (const int*)d_in[18];

    float* outA = (float*)d_out;                      // (1, N, 24)
    float* outI = outA + (size_t)N_ANCHOR * 24;       // (1, N, 256)

    dim3 grid(N_ANCHOR / ANCH_PER_BLOCK), block(THREADS);
    hipLaunchKernelGGL(GaussianNewLifter_kernel, grid, block, 0, stream,
                       anchor, depthnet, depthgt, w2c, cam_k, vox_origin,
                       scene_size, cvr, W1, b1, W2, b2, W3, b3, Wi, bi,
                       f1, f2, outA, outI);
}
